// SSM_5360119186102
// MI455X (gfx1250) — matrix operand order repaired and checked by reading, not compiled, not run on hardware
//
#include <hip/hip_runtime.h>
#include <math.h>

#define BATCH 16
#define HDIM  128   // D_INPUT (= H = O)
#define PDIM  128   // state modes
#define LSEQ  8192

typedef __attribute__((ext_vector_type(16))) __bf16 v16bf;
typedef __attribute__((ext_vector_type(8)))  float  v8f;

__device__ __forceinline__ unsigned int f2bf_bits(float x) {
    union { float f; unsigned int u; } v; v.f = x;
    unsigned int r = v.u + 0x7FFFu + ((v.u >> 16) & 1u);   // round-to-nearest-even
    return r >> 16;
}
__device__ __forceinline__ unsigned short f2bf(float x) {
    return (unsigned short)f2bf_bits(x);
}

// ---------------------------------------------------------------------------
// Kernel 1: ZOH discretization + bf16 packing of all GEMM operand matrices.
// Lambda_bar = exp(Lambda*Delta); B_bar = ((Lambda_bar-1)/Lambda)*B
// Pack: B_bar re/im, Re(C), -Im(C), D as bf16 (row-major 128x128 each),
// plus Lambda_bar re/im f32 vectors for the scan.
// ---------------------------------------------------------------------------
__global__ __launch_bounds__(256) void ssm_setup(
    const float* __restrict__ lam, const float* __restrict__ Bm,
    const float* __restrict__ Cm,  const float* __restrict__ Dm,
    const float* __restrict__ delta,
    unsigned short* __restrict__ bbre, unsigned short* __restrict__ bbim,
    unsigned short* __restrict__ crebf, unsigned short* __restrict__ cimnbf,
    unsigned short* __restrict__ dbf,
    float* __restrict__ lbre, float* __restrict__ lbim)
{
    int idx = blockIdx.x * 256 + threadIdx.x;
    if (idx >= PDIM * HDIM) return;
    int p = idx >> 7;
    int h = idx & 127;

    float la = lam[2 * p], lw = lam[2 * p + 1];
    float dt = delta[p];
    float e  = expf(la * dt);
    float cw = cosf(lw * dt), sw = sinf(lw * dt);
    float lr = e * cw, li = e * sw;                 // Lambda_bar
    float nr = lr - 1.0f, ni = li;                  // Lambda_bar - 1
    float den = la * la + lw * lw;
    float cr = (nr * la + ni * lw) / den;           // (Lb-1)/Lambda
    float ci = (ni * la - nr * lw) / den;

    float br = Bm[2 * idx], bi = Bm[2 * idx + 1];
    bbre[idx] = f2bf(cr * br - ci * bi);
    bbim[idx] = f2bf(cr * bi + ci * br);

    crebf[idx]  = f2bf(Cm[2 * idx]);                // C is (O,P) row-major, same 128x128
    cimnbf[idx] = f2bf(-Cm[2 * idx + 1]);           // pre-negated imag part
    dbf[idx]    = f2bf(Dm[idx]);

    if (h == 0) { lbre[p] = lr; lbim[p] = li; }
}

// ---- WMMA fragment helpers -------------------------------------------------
// A fragment 16x32 bf16 from row-major matrix (row stride 128 bf16).
// Lane = row M=lane&15; K chunks {half*8..+7, 16+half*8..+7} per ISA 7.12.2.
__device__ __forceinline__ v16bf load_a_frag(const unsigned short* __restrict__ mat,
                                             int m0, int k0, int lane)
{
    int mrow = lane & 15, half = lane >> 4;
    const unsigned short* row = mat + (m0 + mrow) * HDIM + k0;
    v16bf a;
    uint4* pa = reinterpret_cast<uint4*>(&a);
    pa[0] = *reinterpret_cast<const uint4*>(row + half * 8);        // K = half*8 .. +7
    pa[1] = *reinterpret_cast<const uint4*>(row + 16 + half * 8);   // K = 16+half*8 .. +7
    return a;
}

// B fragment 32x16 bf16 from LDS tile stored [n][k], pitch 40 bf16.
// Lane = col N=lane&15; K chunks {half*8..+7, 16+half*8..+7} (matches A).
__device__ __forceinline__ v16bf load_b_frag(const unsigned short* __restrict__ lds,
                                             int n0, int lane)
{
    int n = lane & 15, half = lane >> 4;
    const unsigned short* bp = lds + (n0 + n) * 40 + half * 8;
    v16bf b;
    uint4* pb = reinterpret_cast<uint4*>(&b);
    pb[0] = *reinterpret_cast<const uint4*>(bp);
    pb[1] = *reinterpret_cast<const uint4*>(bp + 16);
    return b;
}

// Stage a 32(k) x 64(n) f32 tile -> bf16 LDS [n][k]: each thread converts a
// K-adjacent pair and emits ONE packed ds_store_b32 (half the DS stores of
// per-element b16 staging). Global reads stay coalesced along n (=l).
__device__ __forceinline__ void stage_tile(const float* __restrict__ src,
                                           unsigned int* __restrict__ lds32,
                                           int k0, int l0, int tid)
{
    #pragma unroll
    for (int i = 0; i < 4; ++i) {
        int idx = tid + i * 256;            // 1024 pairs
        int n  = idx & 63;
        int kp = idx >> 6;                  // 0..15 -> k = 2*kp
        const float* g = src + (size_t)(k0 + 2 * kp) * LSEQ + (l0 + n);
        float v0 = g[0];
        float v1 = g[LSEQ];
        lds32[n * 20 + kp] = f2bf_bits(v0) | (f2bf_bits(v1) << 16);  // pitch 40 b16 = 20 u32
    }
}

// ---------------------------------------------------------------------------
// Kernel 2: Bu (complex) = B_bar (PxH) @ u_b (HxL).  Block tile 128(M) x 64(N),
// K=128 in 4 steps of 32. 8 waves: each owns a 16-row strip, 4 N-subtiles,
// dual accumulators (re & im share the staged u tile).
// ---------------------------------------------------------------------------
__global__ __launch_bounds__(256) void ssm_gemm_bu(
    const float* __restrict__ u,
    const unsigned short* __restrict__ bbre, const unsigned short* __restrict__ bbim,
    float* __restrict__ bure, float* __restrict__ buim)
{
    const int l0   = blockIdx.x * 64;
    const int b    = blockIdx.y;
    const int tid  = threadIdx.x;
    const int wave = tid >> 5, lane = tid & 31;
    const int m0   = wave * 16;
    const int half = lane >> 4;

    __shared__ __align__(16) unsigned short lds[64 * 40];
    unsigned int* lds32 = reinterpret_cast<unsigned int*>(lds);

    v8f accr[4], acci[4];
    const v8f vz = {0.f, 0.f, 0.f, 0.f, 0.f, 0.f, 0.f, 0.f};
    for (int nt = 0; nt < 4; ++nt) { accr[nt] = vz; acci[nt] = vz; }

    const float* usrc = u + (size_t)b * HDIM * LSEQ;

    for (int kk = 0; kk < 4; ++kk) {
        const int k0 = kk * 32;
        __syncthreads();
        stage_tile(usrc, lds32, k0, l0, tid);
        __syncthreads();

        // prefetch next K-step's tile rows while WMMAs run (global_prefetch path)
        if (kk < 3) {
            int idx = tid;
            int n = idx & 63, kp = idx >> 6;
            __builtin_prefetch(usrc + (size_t)(k0 + 32 + 2 * kp) * LSEQ + (l0 + n), 0, 3);
        }

        v16bf ar = load_a_frag(bbre, m0, k0, lane);
        v16bf ai = load_a_frag(bbim, m0, k0, lane);
        #pragma unroll
        for (int nt = 0; nt < 4; ++nt) {
            v16bf bf_ = load_b_frag(lds, nt * 16, lane);
            accr[nt] = __builtin_amdgcn_wmma_f32_16x16x32_bf16(
                false, ar, false, bf_, (short)0, accr[nt], false, false);
            acci[nt] = __builtin_amdgcn_wmma_f32_16x16x32_bf16(
                false, ai, false, bf_, (short)0, acci[nt], false, false);
        }
    }

    const size_t obase = (size_t)b * PDIM * LSEQ;
    #pragma unroll
    for (int nt = 0; nt < 4; ++nt)
        #pragma unroll
        for (int r = 0; r < 8; ++r) {
            int row = m0 + half * 8 + r;
            int col = l0 + nt * 16 + (lane & 15);
            bure[obase + (size_t)row * LSEQ + col] = accr[nt][r];
            buim[obase + (size_t)row * LSEQ + col] = acci[nt][r];
        }
}

// ---------------------------------------------------------------------------
// Kernel 3: in-place complex scan x_t = Lb*x_{t-1} + Bu_t along L.
// One wave per (b,p) row. Lane owns 256 contiguous elements; pass 1 computes
// lane-end state, log-step shfl_up prefix of (Lb^256, state) carries, pass 2
// replays with the correct initial state and stores xs in place.
// ---------------------------------------------------------------------------
__global__ __launch_bounds__(256) void ssm_scan(
    float* __restrict__ bure, float* __restrict__ buim,
    const float* __restrict__ lbre, const float* __restrict__ lbim)
{
    const int wid  = blockIdx.x * 8 + (threadIdx.x >> 5);
    const int lane = threadIdx.x & 31;
    const int p    = wid & (PDIM - 1);

    const float ar = lbre[p], ai = lbim[p];
    const size_t base = (size_t)wid * LSEQ + lane * 256;
    const float4* pr = reinterpret_cast<const float4*>(bure + base);
    const float4* pi = reinterpret_cast<const float4*>(buim + base);

    float xr = 0.f, xi = 0.f;
    #define SSM_STEP(RR, II) { float nr_ = ar*xr - ai*xi + (RR); \
                               float ni_ = ar*xi + ai*xr + (II); xr = nr_; xi = ni_; }
    for (int t = 0; t < 64; ++t) {
        float4 r = pr[t], q = pi[t];
        SSM_STEP(r.x, q.x) SSM_STEP(r.y, q.y) SSM_STEP(r.z, q.z) SSM_STEP(r.w, q.w)
    }

    // Lb^256 by 8 squarings
    float aar = ar, aai = ai;
    #pragma unroll
    for (int i = 0; i < 8; ++i) {
        float nr_ = aar * aar - aai * aai;
        float ni_ = 2.f * aar * aai;
        aar = nr_; aai = ni_;
    }

    // inclusive wave scan of (a, s) with combine(prev,cur) = (a_c*a_p, a_c*s_p + s_c)
    float sre = xr, sim = xi, ta = aar, tb = aai;
    #pragma unroll
    for (int off = 1; off < 32; off <<= 1) {
        float par = __shfl_up(ta,  off);
        float pai = __shfl_up(tb,  off);
        float psr = __shfl_up(sre, off);
        float psi = __shfl_up(sim, off);
        if (lane >= off) {
            float nsr = ta * psr - tb * psi + sre;
            float nsi = ta * psi + tb * psr + sim;
            float nar = ta * par - tb * pai;
            float nai = ta * pai + tb * par;
            sre = nsr; sim = nsi; ta = nar; tb = nai;
        }
    }
    float cre = __shfl_up(sre, 1);
    float cim = __shfl_up(sim, 1);
    if (lane == 0) { cre = 0.f; cim = 0.f; }

    // pass 2: replay with carry, store in place
    xr = cre; xi = cim;
    float4* wr = reinterpret_cast<float4*>(bure + base);
    float4* wi = reinterpret_cast<float4*>(buim + base);
    for (int t = 0; t < 64; ++t) {
        float4 r = pr[t], q = pi[t];
        float4 orf, oif;
        SSM_STEP(r.x, q.x) orf.x = xr; oif.x = xi;
        SSM_STEP(r.y, q.y) orf.y = xr; oif.y = xi;
        SSM_STEP(r.z, q.z) orf.z = xr; oif.z = xi;
        SSM_STEP(r.w, q.w) orf.w = xr; oif.w = xi;
        wr[t] = orf; wi[t] = oif;
    }
    #undef SSM_STEP
}

// ---------------------------------------------------------------------------
// Kernel 4: y = Re(C)@xs_re + (-Im(C))@xs_im + D@u, then exact-erf GELU.
// Same tiling as kernel 2; K loop runs over 3 sources x 4 k-steps (K_eff=384),
// one accumulator set. 48 WMMAs per wave.
// ---------------------------------------------------------------------------
__global__ __launch_bounds__(256) void ssm_gemm_out(
    const float* __restrict__ u,
    const float* __restrict__ xsre, const float* __restrict__ xsim,
    const unsigned short* __restrict__ crebf,
    const unsigned short* __restrict__ cimnbf,
    const unsigned short* __restrict__ dbf,
    float* __restrict__ out)
{
    const int l0   = blockIdx.x * 64;
    const int b    = blockIdx.y;
    const int tid  = threadIdx.x;
    const int wave = tid >> 5, lane = tid & 31;
    const int m0   = wave * 16;
    const int half = lane >> 4;

    __shared__ __align__(16) unsigned short lds[64 * 40];
    unsigned int* lds32 = reinterpret_cast<unsigned int*>(lds);

    v8f acc[4];
    const v8f vz = {0.f, 0.f, 0.f, 0.f, 0.f, 0.f, 0.f, 0.f};
    for (int nt = 0; nt < 4; ++nt) acc[nt] = vz;

    const size_t bofs = (size_t)b * HDIM * LSEQ;
    const float* planes[3] = { xsre + bofs, xsim + bofs, u + bofs };
    const unsigned short* amats[3] = { crebf, cimnbf, dbf };

    for (int s = 0; s < 3; ++s) {
        const float* src = planes[s];
        const unsigned short* am = amats[s];
        for (int kk = 0; kk < 4; ++kk) {
            const int k0 = kk * 32;
            __syncthreads();
            stage_tile(src, lds32, k0, l0, tid);
            __syncthreads();

            // prefetch next staged tile (next k-step, or next source plane)
            {
                const float* nsrc = (kk < 3) ? src : (s < 2 ? planes[s + 1] : nullptr);
                int nk0 = (kk < 3) ? (k0 + 32) : 0;
                if (nsrc) {
                    int n = tid & 63, kp = tid >> 6;
                    __builtin_prefetch(nsrc + (size_t)(nk0 + 2 * kp) * LSEQ + (l0 + n), 0, 3);
                }
            }

            v16bf af = load_a_frag(am, m0, k0, lane);
            #pragma unroll
            for (int nt = 0; nt < 4; ++nt) {
                v16bf bf_ = load_b_frag(lds, nt * 16, lane);
                acc[nt] = __builtin_amdgcn_wmma_f32_16x16x32_bf16(
                    false, af, false, bf_, (short)0, acc[nt], false, false);
            }
        }
    }

    const size_t obase = (size_t)b * HDIM * LSEQ;
    #pragma unroll
    for (int nt = 0; nt < 4; ++nt)
        #pragma unroll
        for (int r = 0; r < 8; ++r) {
            int row = m0 + half * 8 + r;
            int col = l0 + nt * 16 + (lane & 15);
            float y = acc[nt][r];
            float g = 0.5f * y * (1.0f + erff(y * 0.70710678118654752f));
            out[obase + (size_t)row * LSEQ + col] = g;
        }
}

// ---------------------------------------------------------------------------
extern "C" void kernel_launch(void* const* d_in, const int* in_sizes, int n_in,
                              void* d_out, int out_size, void* d_ws, size_t ws_size,
                              hipStream_t stream)
{
    (void)in_sizes; (void)n_in; (void)out_size; (void)ws_size;
    const float* u     = (const float*)d_in[0];   // (B,H,L) f32
    const float* lam   = (const float*)d_in[1];   // (P,1) c64 -> interleaved f32
    const float* Bm    = (const float*)d_in[2];   // (P,H) c64
    const float* Cm    = (const float*)d_in[3];   // (O,P) c64
    const float* Dm    = (const float*)d_in[4];   // (O,H) f32
    const float* delta = (const float*)d_in[5];   // (P,1) f32
    float* out = (float*)d_out;

    const size_t plane = (size_t)BATCH * PDIM * LSEQ;
    float* f = (float*)d_ws;
    float* bure = f;                  // Bu_re, becomes xs_re in place
    float* buim = f + plane;          // Bu_im, becomes xs_im in place
    unsigned short* bbre   = (unsigned short*)(f + 2 * plane);
    unsigned short* bbim   = bbre   + PDIM * HDIM;
    unsigned short* crebf  = bbim   + PDIM * HDIM;
    unsigned short* cimnbf = crebf  + PDIM * HDIM;
    unsigned short* dbf    = cimnbf + PDIM * HDIM;
    float* lbre = (float*)(dbf + PDIM * HDIM);
    float* lbim = lbre + PDIM;

    ssm_setup<<<dim3((PDIM * HDIM + 255) / 256), dim3(256), 0, stream>>>(
        lam, Bm, Cm, Dm, delta, bbre, bbim, crebf, cimnbf, dbf, lbre, lbim);

    ssm_gemm_bu<<<dim3(LSEQ / 64, BATCH), dim3(256), 0, stream>>>(
        u, bbre, bbim, bure, buim);

    ssm_scan<<<dim3((BATCH * PDIM) / 8), dim3(256), 0, stream>>>(
        bure, buim, lbre, lbim);

    ssm_gemm_out<<<dim3(LSEQ / 64, BATCH), dim3(256), 0, stream>>>(
        u, bure, buim, crebf, cimnbf, dbf, out);
}
